// LSTM_26912265077001
// MI455X (gfx1250) — matrix operand order repaired and checked by reading, not compiled, not run on hardware
//
#include <hip/hip_runtime.h>

// ---------------------------------------------------------------------------
// Persistent 2-layer LSTM for MI455X (gfx1250, wave32, WMMA).
//
// Latency-bound recurrence (T=512 dependent steps, ~33 GFLOP, <4MB HBM
// traffic -> pure on-chip latency problem). Design:
//   - Weights live in loop-invariant WMMA B-fragments (VGPRs), with gate
//     columns PERMUTED to n = 4*j + gate so each wave owns all 4 gates of
//     4 hidden units -> pointwise is in-register: lane-uniform activation,
//     3x ds_swizzle quad-gather (imm-encoded, no index VGPRs), c in VGPRs.
//   - h state: ping-pong f16 LDS buffers in WMMA-A-friendly layout.
//   - ONE barrier per timestep (h1 handoff). All other cross-step hazards
//     are resolved by the ping-pong buffers; the projection is delayed one
//     step so it reads a stable buffer right after the barrier.
// Grid: 64 WGs (batch tiles of 16). Block: 13 waves (416 thr). Per step:
// 6 dependent v_wmma + 1 barrier.
// ---------------------------------------------------------------------------

namespace {
constexpr int kH  = 51;          // hidden
constexpr int kT  = 512;         // timesteps
constexpr int kB  = 1024;        // batch
constexpr int kMB = 16;          // batch tile per workgroup (WMMA M)
constexpr int kNT = 13;          // N tiles of 16 (13*4 = 52 >= 51 hidden)
constexpr int kK1 = 64;          // padded K for cell1 (h1: 51 -> 64)
constexpr int kK2 = 128;         // padded K for cell2 ([h1|h2]: 102 -> 128)
constexpr int kThreads = kNT * 32;  // 416
}

typedef __attribute__((ext_vector_type(16))) _Float16 v16h;
typedef __attribute__((ext_vector_type(8)))  _Float16 v8h;
typedef __attribute__((ext_vector_type(8)))  float    v8f;

union AV { v16h v; v8h h[2]; };

// group-of-4 ds_swizzle: imm = 0x8000 | {sel3,sel2,sel1,sel0} (2b each)
#define QSWZ_XOR1 0x80B1   // sels (1,0,3,2)
#define QSWZ_XOR2 0x804E   // sels (2,3,0,1)
#define QSWZ_XOR3 0x801B   // sels (3,2,1,0)

__device__ __forceinline__ float qswz(float v, int imm) {
  union { float f; int i; } u;
  u.f = v;
  switch (imm) {   // imm must be literal for the builtin
    case QSWZ_XOR1: u.i = __builtin_amdgcn_ds_swizzle(u.i, QSWZ_XOR1); break;
    case QSWZ_XOR2: u.i = __builtin_amdgcn_ds_swizzle(u.i, QSWZ_XOR2); break;
    default:        u.i = __builtin_amdgcn_ds_swizzle(u.i, QSWZ_XOR3); break;
  }
  return u.f;
}

// sigmoid/tanh via v_exp_f32 + v_rcp_f32 (no IEEE div sequence)
__device__ __forceinline__ float fast_sigmoid(float x) {
  return __builtin_amdgcn_rcpf(1.0f + __expf(-x));
}
__device__ __forceinline__ float fast_tanh(float x) {
  return __builtin_fmaf(2.0f, __builtin_amdgcn_rcpf(1.0f + __expf(-2.0f * x)), -1.0f);
}

__global__ __launch_bounds__(kThreads, 1)
void lstm_persistent(const float* __restrict__ x,     // (T,B,1)
                     const float* __restrict__ Wih1,  // (204,1)
                     const float* __restrict__ Whh1,  // (204,51)
                     const float* __restrict__ bih1,  // (204)
                     const float* __restrict__ bhh1,  // (204)
                     const float* __restrict__ Wih2,  // (204,51)
                     const float* __restrict__ Whh2,  // (204,51)
                     const float* __restrict__ bih2,  // (204)
                     const float* __restrict__ bhh2,  // (204)
                     const float* __restrict__ Wlin,  // (1,51)
                     const float* __restrict__ blin,  // (1)
                     float* __restrict__ out)         // (B,T) row-major
{
  // Ping-pong h buffers (f16, zero-padded K so WMMA K-padding stays exact)
  __shared__ __attribute__((aligned(16))) _Float16 h1pp[2][kMB * kK1]; // cell1 A
  __shared__ __attribute__((aligned(16))) _Float16 a2pp[2][kMB * kK2]; // [h1|h2] cell2 A
  __shared__ float wlin_s[52];        // padded: [51] = 0

  const int tid  = threadIdx.x;
  const int lane = tid & 31;
  const int wave = tid >> 5;
  const int nloc = lane & 15;       // col in this wave's N tile / A row M
  const int hi   = lane >> 4;       // lane-half: K sub-range / rows M+8
  const int bb   = blockIdx.x * kMB;

  // Interleaved gate-column permutation: tile col n -> (hidden j, gate g)
  const int gate = nloc & 3;                 // 0:i 1:f 2:g 3:o
  const int j    = wave * 4 + (nloc >> 2);   // hidden unit 0..51 (51 = pad)
  const bool jv  = (j < kH);
  const int grow = gate * kH + j;            // PyTorch gate row index

  // ---- zero-init recurrent LDS state (padding stays zero forever) ----
  for (int i = tid; i < 2 * kMB * kK1; i += kThreads) h1pp[0][i] = (_Float16)0.f;
  for (int i = tid; i < 2 * kMB * kK2; i += kThreads) a2pp[0][i] = (_Float16)0.f;
  for (int i = tid; i < 52; i += kThreads) wlin_s[i] = (i < kH) ? Wlin[i] : 0.f;

  // ---- per-lane loop-invariant scalars ----
  const float bias1 = jv ? (bih1[grow] + bhh1[grow]) : 0.f;
  const float bias2 = jv ? (bih2[grow] + bhh2[grow]) : 0.f;
  const float wx    = jv ? Wih1[grow] : 0.f;   // IN==1: x term is rank-1
  const float b0    = blin[0];
  // lane-uniform activation: act(v) = sA * sigmoid(scl*v) + sB
  const bool  isT = (gate == 2);
  const float scl = isT ? 2.f : 1.f;
  const float sA  = isT ? 2.f : 1.f;
  const float sB  = isT ? -1.f : 0.f;

  // ---- loop-invariant WMMA B fragments (permuted columns), from global ----
  v16h B1[2], B2[4];
  #pragma unroll
  for (int kt = 0; kt < 2; ++kt)
    #pragma unroll
    for (int e = 0; e < 16; ++e) {
      const int k = kt * 32 + hi * 16 + e;
      B1[kt][e] = (_Float16)((jv && k < kH) ? Whh1[grow * kH + k] : 0.f);
    }
  #pragma unroll
  for (int kt = 0; kt < 4; ++kt)
    #pragma unroll
    for (int e = 0; e < 16; ++e) {
      const int k = kt * 32 + hi * 16 + e;
      float v = 0.f;
      if (jv) {
        if (k < kH)                      v = Wih2[grow * kH + k];
        else if (k >= 64 && k < 64 + kH) v = Whh2[grow * kH + (k - 64)];
      }
      B2[kt][e] = (_Float16)v;
    }

  // ---- register-resident cell state (valid in the gate==0 lane) ----
  float c1r[8], c2r[8];
  #pragma unroll
  for (int r = 0; r < 8; ++r) { c1r[r] = 0.f; c2r[r] = 0.f; }

  // ---- x prefetch registers ----
  const float* xbase = x + bb + hi * 8;
  float xr[8];
  #pragma unroll
  for (int r = 0; r < 8; ++r) xr[r] = xbase[r];

  _Float16* h1rd = h1pp[0]; _Float16* h1wr = h1pp[1];
  _Float16* a2rd = a2pp[0]; _Float16* a2wr = a2pp[1];

  __syncthreads();

  // =========================== recurrent T loop ===========================
  #pragma unroll 1
  for (int t = 0; t < kT; ++t) {
    // ---- cell 1 GEMM: acc = bias + x*wih1 + h1 @ Whh1^T (permuted N) ----
    v8f acc;
    #pragma unroll
    for (int r = 0; r < 8; ++r) acc[r] = __builtin_fmaf(xr[r], wx, bias1);

    // prefetch x(t+1); latency hides under both cells
    {
      const float* xnp = xbase + (size_t)((t + 1 < kT) ? (t + 1) : t) * kB;
      #pragma unroll
      for (int r = 0; r < 8; ++r) xr[r] = xnp[r];
    }

    {
      const _Float16* row = h1rd + nloc * kK1;
      AV a0, a1;
      a0.h[0] = *(const v8h*)(row +      hi * 16);
      a0.h[1] = *(const v8h*)(row +  8 + hi * 16);
      a1.h[0] = *(const v8h*)(row + 32 + hi * 16);
      a1.h[1] = *(const v8h*)(row + 40 + hi * 16);
      acc = __builtin_amdgcn_wmma_f32_16x16x32_f16(false, a0.v, false, B1[0], (short)0, acc, false, false);
      acc = __builtin_amdgcn_wmma_f32_16x16x32_f16(false, a1.v, false, B1[1], (short)0, acc, false, false);
    }

    // ---- cell 1 pointwise, in-register (gate==0 lane computes) ----
    {
      float av[8], hv[8];
      #pragma unroll
      for (int r = 0; r < 8; ++r)
        av[r] = __builtin_fmaf(sA, fast_sigmoid(scl * acc[r]), sB);
      #pragma unroll
      for (int r = 0; r < 8; ++r) {
        const float f_ = qswz(av[r], QSWZ_XOR1);  // lane g0 <- f
        const float g_ = qswz(av[r], QSWZ_XOR2);  // lane g0 <- g
        const float o_ = qswz(av[r], QSWZ_XOR3);  // lane g0 <- o
        const float cn = __builtin_fmaf(f_, c1r[r], av[r] * g_); // av = i in g0
        c1r[r] = cn;
        hv[r] = o_ * fast_tanh(cn);
      }
      if (gate == 0 && jv) {          // one writer per 4-lane group
        #pragma unroll
        for (int r = 0; r < 8; ++r) {
          const int m = r + hi * 8;
          const _Float16 hh = (_Float16)hv[r];
          h1wr[m * kK1 + j] = hh;       // next step's cell-1 A
          a2rd[m * kK2 + j] = hh;       // this step's cell-2 A (K 0..50)
        }
      }
    }
    __syncthreads();   // the ONLY barrier: h1 visible before cell-2 GEMM

    // ---- delayed projection: out[b][t-1] = h2(t-1) . Wlin + b ----
    // h2(t-1) sits in a2rd's h2 region: written pre-barrier at t-1, and
    // pw2(t) writes the OTHER buffer, so it is stable here. Wave 0 only;
    // each row split across 2 lanes (26-FMA chains), combined via shfl_xor.
    if (t > 0 && tid < 32) {
      const int m = lane & 15, half = lane >> 4;
      const _Float16* hr = a2rd + m * kK2 + 64 + half * 26;
      const float*    wr = wlin_s + half * 26;
      float s = half ? 0.f : b0;
      #pragma unroll
      for (int jj = 0; jj < 26; ++jj)
        s = __builtin_fmaf((float)hr[jj], wr[jj], s);
      s += __shfl_xor(s, 16, 32);
      if (half == 0) out[(size_t)(bb + m) * kT + (t - 1)] = s;
    }

    // ---- cell 2 GEMM: acc2 = bias2 + [h1|h2] @ [Wih2|Whh2]^T ----
    v8f acc2;
    #pragma unroll
    for (int r = 0; r < 8; ++r) acc2[r] = bias2;
    {
      const _Float16* row = a2rd + nloc * kK2;
      #pragma unroll
      for (int kt = 0; kt < 4; ++kt) {
        AV a;
        a.h[0] = *(const v8h*)(row + kt * 32 +     hi * 16);
        a.h[1] = *(const v8h*)(row + kt * 32 + 8 + hi * 16);
        acc2 = __builtin_amdgcn_wmma_f32_16x16x32_f16(false, a.v, false, B2[kt], (short)0, acc2, false, false);
      }
    }

    // ---- cell 2 pointwise, in-register ----
    {
      float av[8], hv[8];
      #pragma unroll
      for (int r = 0; r < 8; ++r)
        av[r] = __builtin_fmaf(sA, fast_sigmoid(scl * acc2[r]), sB);
      #pragma unroll
      for (int r = 0; r < 8; ++r) {
        const float f_ = qswz(av[r], QSWZ_XOR1);
        const float g_ = qswz(av[r], QSWZ_XOR2);
        const float o_ = qswz(av[r], QSWZ_XOR3);
        const float cn = __builtin_fmaf(f_, c2r[r], av[r] * g_);
        c2r[r] = cn;
        hv[r] = o_ * fast_tanh(cn);
      }
      if (gate == 0 && jv) {
        #pragma unroll
        for (int r = 0; r < 8; ++r) {
          const int m = r + hi * 8;
          a2wr[m * kK2 + 64 + j] = (_Float16)hv[r]; // next step's h2 (K 64..114)
        }
      }
    }

    // swap ping-pong buffers (no end-of-step barrier needed: all remaining
    // cross-wave orderings are covered by next iteration's barrier + pp)
    _Float16* tr = h1rd; h1rd = h1wr; h1wr = tr;
    tr = a2rd; a2rd = a2wr; a2wr = tr;
  }

  // ---- flush last projection column (t = kT-1) ----
  __syncthreads();
  if (tid < 32) {
    const int m = lane & 15, half = lane >> 4;
    const _Float16* hr = a2rd + m * kK2 + 64 + half * 26;  // holds h2(kT-1)
    const float*    wr = wlin_s + half * 26;
    float s = half ? 0.f : b0;
    #pragma unroll
    for (int jj = 0; jj < 26; ++jj)
      s = __builtin_fmaf((float)hr[jj], wr[jj], s);
    s += __shfl_xor(s, 16, 32);
    if (half == 0) out[(size_t)(bb + m) * kT + (kT - 1)] = s;
  }
}

extern "C" void kernel_launch(void* const* d_in, const int* in_sizes, int n_in,
                              void* d_out, int out_size, void* d_ws, size_t ws_size,
                              hipStream_t stream) {
  const float* x    = (const float*)d_in[0];
  const float* Wih1 = (const float*)d_in[1];
  const float* Whh1 = (const float*)d_in[2];
  const float* bih1 = (const float*)d_in[3];
  const float* bhh1 = (const float*)d_in[4];
  const float* Wih2 = (const float*)d_in[5];
  const float* Whh2 = (const float*)d_in[6];
  const float* bih2 = (const float*)d_in[7];
  const float* bhh2 = (const float*)d_in[8];
  const float* Wlin = (const float*)d_in[9];
  const float* blin = (const float*)d_in[10];
  // d_in[11] = future (static 0) -> ignored
  float* out = (float*)d_out;

  dim3 grid(kB / kMB);      // 64 workgroups, one batch tile of 16 each
  dim3 block(kThreads);     // 13 waves: one per 16-wide (permuted) gate tile
  lstm_persistent<<<grid, block, 0, stream>>>(
      x, Wih1, Whh1, bih1, bhh1, Wih2, Whh2, bih2, bhh2, Wlin, blin, out);
}
